// MolFPKG_DILI_42992622633396
// MI455X (gfx1250) — compile-verified, matrix operand order checked by reading, not run on hardware
//
#include <hip/hip_runtime.h>
#include <hip/hip_bf16.h>

#define N_DRUGS 4096
#define N_GENE  5414
#define N_KG    (N_DRUGS + N_GENE)   // 9510
#define N_MOL   131072
#define E_MOL   524288
#define E_KG    524288
#define R_REL   8
#define MLP_IN  1024
#define MLP_H   512
#define DIM_D   256
#define GCN_IN  64
#define GCN_H   128
#define DIM1    256
#define DIM2    128
#define DIM3    64
#define EPS_BN  1e-5f

typedef __attribute__((ext_vector_type(16))) __bf16 v16bf;
typedef __attribute__((ext_vector_type(8)))  float  v8f;

__device__ __forceinline__ unsigned int pack2bf(float f0, float f1) {
    unsigned int u0 = __builtin_bit_cast(unsigned int, f0);
    unsigned int u1 = __builtin_bit_cast(unsigned int, f1);
    u0 = u0 + 0x7FFFu + ((u0 >> 16) & 1u);          // RNE
    u1 = u1 + 0x7FFFu + ((u1 >> 16) & 1u);
    return (u0 >> 16) | (u1 & 0xFFFF0000u);
}

// ---------------------------------------------------------------------------
// Weight pre-pass: B[K,N] f32 -> Bt[N,K] bf16 bits (batched via blockIdx.y)
// ---------------------------------------------------------------------------
__global__ __launch_bounds__(256) void transpose_bf16(const float* __restrict__ B,
                                                      unsigned short* __restrict__ Bt,
                                                      int K, int N)
{
    long long base = (long long)blockIdx.y * K * N;
    long long idx  = (long long)blockIdx.x * 256 + threadIdx.x;
    if (idx >= (long long)K * N) return;
    int k = (int)(idx / N), n = (int)(idx % N);
    unsigned int u = __builtin_bit_cast(unsigned int, B[base + idx]);
    u = u + 0x7FFFu + ((u >> 16) & 1u);
    Bt[base + (long long)n * K + k] = (unsigned short)(u >> 16);
}

// ---------------------------------------------------------------------------
// WMMA GEMM: C[M,N] = A[M,K] @ B[K,N].  A row-major f32; B supplied transposed
// as bf16 Bt[N,K].  Block tile 64x64, BK=64, 8 waves -> wave = 16 rows x 32
// cols (2 accumulators, 4 WMMA per K-tile).  K % 64 == 0, N % 64 == 0,
// M may be ragged.  blockIdx.z batches via strides (elements).
// B tile staged global->LDS with GLOBAL_LOAD_ASYNC_TO_LDS_B128 (ASYNCcnt);
// A tile converted f32->bf16 through VGPRs; next A tile prefetched.
// ---------------------------------------------------------------------------
#define BM 64
#define BN 64
#define BKK 64
#define LDSP 72   // row pitch in bf16 elems: 144B, 16B-aligned, bank-skewed

__global__ __launch_bounds__(256) void gemm_wmma_bf16(
    const float* __restrict__ A, const unsigned short* __restrict__ Bt,
    float* __restrict__ C, int M, int N, int K,
    long long sA, long long sB, long long sC)
{
    __shared__ unsigned short As[BM * LDSP];
    __shared__ unsigned short Bs[BN * LDSP];
    A  += (long long)blockIdx.z * sA;
    Bt += (long long)blockIdx.z * sB;
    C  += (long long)blockIdx.z * sC;

    const int t    = threadIdx.x;
    const int lane = t & 31;
    const int wave = t >> 5;        // 0..7
    const int wm   = wave >> 1;     // 0..3
    const int wn   = wave & 1;      // 0..1
    const int rowBase = blockIdx.x * BM;
    const int colBase = blockIdx.y * BN;
    const int lhalf = lane >> 4;
    const int l15   = lane & 15;

    // staging: thread t owns row/col ar = t>>2, k-quad akq = (t&3)*16
    const int ar  = t >> 2;
    const int akq = (t & 3) * 16;
    const int gr  = rowBase + ar;
    const bool inb   = (gr < M);
    const bool fullM = (rowBase + BM <= M);
    const float* Arow          = A  + (long long)gr * K;
    const unsigned short* Brow = Bt + (long long)(colBase + ar) * K;

    // LDS byte address for this thread's B-tile slot (generic ptr low 32 bits
    // are the LDS offset on gfx1250)
    const unsigned bs_lds = (unsigned)(size_t)&Bs[ar * LDSP + akq];

    v8f acc0 = {}; v8f acc1 = {};

    for (int kk = 0; kk < K; kk += BKK) {
        // ---- stage B: async DMA global->LDS, 2 x 16B (offset applies to
        //      both global and LDS sides per ISA) ----
        {
            const unsigned short* g = Brow + kk;
            asm volatile("global_load_async_to_lds_b128 %0, %1, off"
                         :: "v"(bs_lds), "v"(g) : "memory");
            asm volatile("global_load_async_to_lds_b128 %0, %1, off offset:16"
                         :: "v"(bs_lds), "v"(g) : "memory");
        }
        // ---- stage A: 4x float4 load -> bf16 pack -> 2x b128 LDS store ----
        uint4 p0 = {0,0,0,0}, p1 = {0,0,0,0};
        if (fullM || inb) {
            const float4* src = (const float4*)(Arow + kk + akq);
            float4 f0 = src[0], f1 = src[1], f2 = src[2], f3 = src[3];
            p0.x = pack2bf(f0.x, f0.y); p0.y = pack2bf(f0.z, f0.w);
            p0.z = pack2bf(f1.x, f1.y); p0.w = pack2bf(f1.z, f1.w);
            p1.x = pack2bf(f2.x, f2.y); p1.y = pack2bf(f2.z, f2.w);
            p1.z = pack2bf(f3.x, f3.y); p1.w = pack2bf(f3.z, f3.w);
            if (kk + BKK < K) __builtin_prefetch(Arow + kk + BKK + akq, 0, 1);
        }
        *(uint4*)&As[ar * LDSP + akq]     = p0;
        *(uint4*)&As[ar * LDSP + akq + 8] = p1;

        asm volatile("s_wait_asynccnt 0" ::: "memory");
        __syncthreads();

        // fragments per ISA 16-bit layout: lane<16 -> K{0..7,16..23},
        // lane>=16 -> K{8..15,24..31} within each 32-k step.
        const int am  = wm * 16 + l15;
        const int bn0 = wn * 32 + l15;
        union Frag { uint4 q[2]; v16bf v; };
        #pragma unroll
        for (int s = 0; s < 2; ++s) {
            const int kb = s * 32 + lhalf * 8;
            Frag af, b0, b1;
            af.q[0] = *(const uint4*)&As[am * LDSP + kb];
            af.q[1] = *(const uint4*)&As[am * LDSP + kb + 16];
            b0.q[0] = *(const uint4*)&Bs[bn0 * LDSP + kb];
            b0.q[1] = *(const uint4*)&Bs[bn0 * LDSP + kb + 16];
            b1.q[0] = *(const uint4*)&Bs[(bn0 + 16) * LDSP + kb];
            b1.q[1] = *(const uint4*)&Bs[(bn0 + 16) * LDSP + kb + 16];
            acc0 = __builtin_amdgcn_wmma_f32_16x16x32_bf16(false, af.v, false, b0.v,
                                                           (short)0, acc0, false, false);
            acc1 = __builtin_amdgcn_wmma_f32_16x16x32_bf16(false, af.v, false, b1.v,
                                                           (short)0, acc1, false, false);
        }
        __syncthreads();
    }

    const int rb = rowBase + wm * 16 + lhalf * 8;
    const int cb = colBase + wn * 32 + l15;
    #pragma unroll
    for (int j = 0; j < 8; ++j) {
        int r = rb + j;
        if (r < M) {
            C[(long long)r * N + cb]      = acc0[j];
            C[(long long)r * N + cb + 16] = acc1[j];
        }
    }
}

// ---------------------------------------------------------------------------
// BatchNorm (batch stats) + ReLU — coalesced two-stage column stats
// ---------------------------------------------------------------------------
__global__ __launch_bounds__(256) void col_partial(const float* __restrict__ X, int M, int N,
                                                   float* __restrict__ sum, float* __restrict__ ssq)
{
    __shared__ float s1[256], s2[256];
    int c   = blockIdx.x * 64 + (threadIdx.x & 63);
    int rlo = blockIdx.y * 256;
    int rhi = rlo + 256; if (rhi > M) rhi = M;
    float a = 0.f, b = 0.f;
    for (int r = rlo + (threadIdx.x >> 6); r < rhi; r += 4) {
        float v = X[(long long)r * N + c];
        a += v; b += v * v;
    }
    s1[threadIdx.x] = a; s2[threadIdx.x] = b;
    __syncthreads();
    if (threadIdx.x < 64) {
        a = s1[threadIdx.x] + s1[threadIdx.x + 64] + s1[threadIdx.x + 128] + s1[threadIdx.x + 192];
        b = s2[threadIdx.x] + s2[threadIdx.x + 64] + s2[threadIdx.x + 128] + s2[threadIdx.x + 192];
        atomicAdd(&sum[c], a);
        atomicAdd(&ssq[c], b);
    }
}
__global__ void stats_final(const float* sum, const float* ssq,
                            float* mean, float* rinv, int M, int N)
{
    int c = blockIdx.x * 256 + threadIdx.x;
    if (c >= N) return;
    float m = sum[c] / (float)M;
    float v = ssq[c] / (float)M - m * m;
    mean[c] = m; rinv[c] = rsqrtf(v + EPS_BN);
}
__global__ __launch_bounds__(256) void bn_relu(float* __restrict__ X, long long total, int N,
                                               const float* __restrict__ mean,
                                               const float* __restrict__ rinv)
{
    long long i = (long long)blockIdx.x * 256 + threadIdx.x;
    if (i >= total) return;
    int c = (int)(i % N);
    float v = (X[i] - mean[c]) * rinv[c];
    X[i] = v > 0.f ? v : 0.f;
}

// ---------------------------------------------------------------------------
// small utility kernels
// ---------------------------------------------------------------------------
__global__ void fill_f32(float* p, long long n, float val) {
    long long i = (long long)blockIdx.x * 256 + threadIdx.x;
    if (i < n) p[i] = val;
}
__global__ void vec_add(float* a, const float* b, long long n) {
    long long i = (long long)blockIdx.x * 256 + threadIdx.x;
    if (i < n) a[i] += b[i];
}
__global__ void copy_f32(float* dst, const float* src, long long n) {
    long long i = (long long)blockIdx.x * 256 + threadIdx.x;
    if (i < n) dst[i] = src[i];
}

// ---------------------------------------------------------------------------
// GCN (mol graph): symmetric-normalized scatter with self loops
// ---------------------------------------------------------------------------
__global__ void edge_deg(const int* __restrict__ dst, int E, float* __restrict__ deg) {
    int e = blockIdx.x * 256 + threadIdx.x;
    if (e < E) atomicAdd(&deg[dst[e]], 1.0f);
}
__global__ void deg_to_dinv(float* deg, int n) {
    int i = blockIdx.x * 256 + threadIdx.x;
    if (i < n) deg[i] = rsqrtf(fmaxf(deg[i], 1.0f));
}
__global__ void gcn_self(const float* __restrict__ H, const float* __restrict__ dinv,
                         float* __restrict__ agg, int Nn, int F)
{
    long long i = (long long)blockIdx.x * 256 + threadIdx.x;
    if (i >= (long long)Nn * F) return;
    int row = (int)(i / F);
    float d = dinv[row];
    agg[i] = H[i] * d * d;
}
__global__ __launch_bounds__(256) void gcn_edge(const int* __restrict__ src, const int* __restrict__ dst,
                                                int E, const float* __restrict__ H,
                                                const float* __restrict__ dinv,
                                                float* __restrict__ agg, int F)
{
    int e = blockIdx.x * 8 + (threadIdx.x >> 5);
    if (e >= E) return;
    int lane = threadIdx.x & 31;
    int s = src[e], d = dst[e];
    float nrm = dinv[s] * dinv[d];
    const float* hs = H + (long long)s * F;
    float* ad = agg + (long long)d * F;
    for (int f = lane; f < F; f += 32) atomicAdd(&ad[f], hs[f] * nrm);
}

// ---------------------------------------------------------------------------
// global mean pool per molecule
// ---------------------------------------------------------------------------
__global__ __launch_bounds__(256) void pool_scatter(const float* __restrict__ G,
                                                    const int* __restrict__ batch,
                                                    float* __restrict__ gsum, float* __restrict__ gcnt,
                                                    int Nn, int F)
{
    int i = blockIdx.x * 8 + (threadIdx.x >> 5);
    if (i >= Nn) return;
    int lane = threadIdx.x & 31;
    int b = batch[i];
    const float* g = G + (long long)i * F;
    float* s = gsum + (long long)b * F;
    for (int f = lane; f < F; f += 32) atomicAdd(&s[f], g[f]);
    if (lane == 0) atomicAdd(&gcnt[b], 1.0f);
}
__global__ void pool_div(float* gsum, const float* gcnt, int Nd, int F) {
    long long i = (long long)blockIdx.x * 256 + threadIdx.x;
    if (i >= (long long)Nd * F) return;
    gsum[i] = gsum[i] / fmaxf(gcnt[i / F], 1.0f);
}

// ---------------------------------------------------------------------------
// attention fusion
// ---------------------------------------------------------------------------
__global__ __launch_bounds__(128) void att_score(const float* __restrict__ graph_out,
                                                 const float* __restrict__ fp,
                                                 const float* __restrict__ w1,
                                                 const float* __restrict__ b1,
                                                 const float* __restrict__ w2,
                                                 float* __restrict__ attw)
{
    int n = blockIdx.x, v = blockIdx.y;
    const float* z = (v == 0 ? graph_out : fp) + (long long)n * DIM_D;
    __shared__ float zs[DIM_D];
    __shared__ float red[128];
    int j = threadIdx.x;
    zs[j] = z[j]; zs[j + 128] = z[j + 128];
    __syncthreads();
    float acc = 0.f;
    for (int d = 0; d < DIM_D; ++d) acc += zs[d] * w1[d * 128 + j];
    red[j] = tanhf(acc + b1[j]) * w2[j];
    __syncthreads();
    for (int s = 64; s > 0; s >>= 1) { if (j < s) red[j] += red[j + s]; __syncthreads(); }
    if (j == 0) attw[n * 2 + v] = red[0];
}

__global__ __launch_bounds__(256) void att_fuse(const float* __restrict__ graph_out,
                                                const float* __restrict__ fp,
                                                const float* __restrict__ attw,
                                                float* __restrict__ X, float* __restrict__ beta_out)
{
    int n = blockIdx.x, d = threadIdx.x;
    float w0 = attw[n * 2], w1 = attw[n * 2 + 1];
    float m = fmaxf(w0, w1);
    float e0 = expf(w0 - m), e1 = expf(w1 - m);
    float inv = 1.0f / (e0 + e1);
    float b0 = e0 * inv, b1 = e1 * inv;
    X[(long long)n * DIM_D + d] = b0 * graph_out[(long long)n * DIM_D + d]
                                + b1 * fp[(long long)n * DIM_D + d];
    if (d == 0) { beta_out[n * 2] = b0; beta_out[n * 2 + 1] = b1; }
}

// ---------------------------------------------------------------------------
// RGCN scatter (mean per relation)
// ---------------------------------------------------------------------------
__global__ void rgcn_cnt(const int* __restrict__ dst, const int* __restrict__ etype,
                         int E, float* __restrict__ cnt)
{
    int e = blockIdx.x * 256 + threadIdx.x;
    if (e < E) atomicAdd(&cnt[etype[e] * N_KG + dst[e]], 1.0f);
}
__global__ __launch_bounds__(256) void rgcn_edge(const int* __restrict__ src, const int* __restrict__ dst,
                                                 const int* __restrict__ etype, int E,
                                                 const float* __restrict__ HR,
                                                 const float* __restrict__ cnt,
                                                 float* __restrict__ agg, int F)
{
    int e = blockIdx.x * 8 + (threadIdx.x >> 5);
    if (e >= E) return;
    int lane = threadIdx.x & 31;
    int r = etype[e], s = src[e], d = dst[e];
    float nrm = 1.0f / fmaxf(cnt[r * N_KG + d], 1.0f);
    const float* h = HR + ((long long)r * N_KG + s) * F;
    float* a = agg + (long long)d * F;
    for (int f = lane; f < F; f += 32) atomicAdd(&a[f], h[f] * nrm);
}

// ---------------------------------------------------------------------------
// classifier head: logits (64->2) + log_softmax
// ---------------------------------------------------------------------------
__global__ void lin2_logsoftmax(const float* __restrict__ X3, const float* __restrict__ W,
                                const float* __restrict__ b, float* __restrict__ out, int M)
{
    int r = blockIdx.x * 256 + threadIdx.x;
    if (r >= M) return;
    float l0 = b[0], l1 = b[1];
    const float* x = X3 + (long long)r * DIM3;
    for (int k = 0; k < DIM3; ++k) { float xv = x[k]; l0 += xv * W[k * 2]; l1 += xv * W[k * 2 + 1]; }
    float m = fmaxf(l0, l1);
    float lse = m + logf(expf(l0 - m) + expf(l1 - m));
    out[r * 2] = l0 - lse; out[r * 2 + 1] = l1 - lse;
}

// ---------------------------------------------------------------------------
static inline unsigned div_up(long long a, long long b) { return (unsigned)((a + b - 1) / b); }

extern "C" void kernel_launch(void* const* d_in, const int* in_sizes, int n_in,
                              void* d_out, int out_size, void* d_ws, size_t ws_size,
                              hipStream_t stream)
{
    (void)in_sizes; (void)n_in; (void)out_size; (void)ws_size;
    const float* fp_data  = (const float*)d_in[0];
    const float* mol_x    = (const float*)d_in[1];
    const int*   mol_bat  = (const int*)  d_in[2];
    const int*   mol_ei   = (const int*)  d_in[3];   // [2, E_MOL]
    const int*   kg_ei    = (const int*)  d_in[4];   // [2, E_KG]
    const int*   kg_et    = (const int*)  d_in[5];
    const float* fp_w1    = (const float*)d_in[6];
    const float* fp_w2    = (const float*)d_in[8];
    const float* gcn_w1   = (const float*)d_in[10];
    const float* gcn_w2   = (const float*)d_in[12];
    const float* att_w1   = (const float*)d_in[14];
    const float* att_b1   = (const float*)d_in[15];
    const float* att_w2   = (const float*)d_in[16];
    const float* gene_emb = (const float*)d_in[17];
    const float* rg_w1    = (const float*)d_in[18];
    const float* rg_root1 = (const float*)d_in[19];
    const float* rg_w2    = (const float*)d_in[21];
    const float* rg_root2 = (const float*)d_in[22];
    const float* lin1_w   = (const float*)d_in[24];
    const float* lin2_w   = (const float*)d_in[26];
    const float* lin2_b   = (const float*)d_in[27];
    float* out = (float*)d_out;        // [9510*2 logprobs | 4096*2 beta]

    const int* msrc = mol_ei;           const int* mdst = mol_ei + E_MOL;
    const int* ksrc = kg_ei;            const int* kdst = kg_ei + E_KG;

    // ---- workspace layout (floats) ----
    float* w = (float*)d_ws;
    size_t off = 0;
    float* A1    = w + off; off += (size_t)N_DRUGS * MLP_H;      // 4096x512
    float* FP    = w + off; off += (size_t)N_DRUGS * DIM_D;      // 4096x256
    float* H1    = w + off; off += (size_t)N_MOL * GCN_H;        // 131072x128
    float* AGG1  = w + off; off += (size_t)N_MOL * GCN_H;        // -> G1
    float* H2    = w + off; off += (size_t)N_MOL * DIM_D;        // 131072x256
    float* AGG2  = w + off; off += (size_t)N_MOL * DIM_D;        // -> G2
    float* DEG   = w + off; off += (size_t)N_MOL;
    float* GSUM  = w + off; off += (size_t)N_DRUGS * DIM_D;      // -> graph_out
    float* GCNT  = w + off; off += (size_t)N_DRUGS;
    float* ATTW  = w + off; off += (size_t)N_DRUGS * 2;
    float* X     = w + off; off += (size_t)N_KG * DIM_D;         // 9510x256
    float* CNT   = w + off; off += (size_t)R_REL * N_KG;
    float* AGGK  = w + off; off += (size_t)N_KG * DIM1;          // -> X1
    float* XROOT = w + off; off += (size_t)N_KG * DIM1;          // reused in layer 2
    float* X3    = w + off; off += (size_t)N_KG * DIM3;
    float* MEAN  = w + off; off += 512;
    float* RINV  = w + off; off += 512;
    float* SUM   = w + off; off += 512;
    float* SSQ   = w + off; off += 512;
    // bf16 transposed weights region (16B aligned)
    off = (off + 3) & ~(size_t)3;
    unsigned short* ub = (unsigned short*)(w + off);
    size_t uo = 0;
    unsigned short* FPW1T = ub + uo; uo += (size_t)MLP_IN * MLP_H;   // 1024x512
    unsigned short* FPW2T = ub + uo; uo += (size_t)MLP_H * DIM_D;
    unsigned short* GW1T  = ub + uo; uo += (size_t)GCN_IN * GCN_H;
    unsigned short* GW2T  = ub + uo; uo += (size_t)GCN_H * DIM_D;
    unsigned short* RGW1T = ub + uo; uo += (size_t)R_REL * DIM_D * DIM1;
    unsigned short* RGR1T = ub + uo; uo += (size_t)DIM_D * DIM1;
    unsigned short* RGW2T = ub + uo; uo += (size_t)R_REL * DIM1 * DIM2;
    unsigned short* RGR2T = ub + uo; uo += (size_t)DIM1 * DIM2;
    unsigned short* L1WT  = ub + uo; uo += (size_t)DIM2 * DIM3;
    // aliases (lifetimes disjoint): HR spans H1+AGG1; HR2 spans H2
    float* HR    = H1;      // 8x9510x256 = 19.5M <= 33.5M (H1+AGG1)
    float* HR2   = H2;      // 8x9510x128 =  9.7M <= 33.5M
    float* AGGK2 = XROOT;                          // 9510x128
    float* XROOT2= XROOT + (size_t)N_KG * DIM2;    // 9510x128

    auto xpose = [&](const float* B, unsigned short* Bt, int K, int N, int batch) {
        dim3 g(div_up((long long)K * N, 256), batch);
        transpose_bf16<<<g, 256, 0, stream>>>(B, Bt, K, N);
    };
    auto gemm = [&](const float* A, const unsigned short* Bt, float* C, int M, int N, int K,
                    int batch, long long sA, long long sB, long long sC) {
        dim3 g(div_up(M, BM), N / BN, batch);
        gemm_wmma_bf16<<<g, 256, 0, stream>>>(A, Bt, C, M, N, K, sA, sB, sC);
    };
    auto bn_relu_seq = [&](float* Xp, int M, int N) {
        hipMemsetAsync(SUM, 0, N * sizeof(float), stream);
        hipMemsetAsync(SSQ, 0, N * sizeof(float), stream);
        dim3 g(N / 64, div_up(M, 256));
        col_partial<<<g, 256, 0, stream>>>(Xp, M, N, SUM, SSQ);
        stats_final<<<div_up(N, 256), 256, 0, stream>>>(SUM, SSQ, MEAN, RINV, M, N);
        long long tot = (long long)M * N;
        bn_relu<<<div_up(tot, 256), 256, 0, stream>>>(Xp, tot, N, MEAN, RINV);
    };

    // ===== weight pre-pass: transpose + bf16 convert (tiny) =====
    xpose(fp_w1,    FPW1T, MLP_IN, MLP_H, 1);
    xpose(fp_w2,    FPW2T, MLP_H,  DIM_D, 1);
    xpose(gcn_w1,   GW1T,  GCN_IN, GCN_H, 1);
    xpose(gcn_w2,   GW2T,  GCN_H,  DIM_D, 1);
    xpose(rg_w1,    RGW1T, DIM_D,  DIM1,  R_REL);
    xpose(rg_root1, RGR1T, DIM_D,  DIM1,  1);
    xpose(rg_w2,    RGW2T, DIM1,   DIM2,  R_REL);
    xpose(rg_root2, RGR2T, DIM1,   DIM2,  1);
    xpose(lin1_w,   L1WT,  DIM2,   DIM3,  1);

    // ===== fingerprint MLP (biases before BN wash out) =====
    gemm(fp_data, FPW1T, A1, N_DRUGS, MLP_H, MLP_IN, 1, 0, 0, 0);
    bn_relu_seq(A1, N_DRUGS, MLP_H);
    gemm(A1, FPW2T, FP, N_DRUGS, DIM_D, MLP_H, 1, 0, 0, 0);
    bn_relu_seq(FP, N_DRUGS, DIM_D);

    // ===== molecular GCN =====
    fill_f32<<<div_up(N_MOL, 256), 256, 0, stream>>>(DEG, N_MOL, 1.0f);   // self loop
    edge_deg<<<div_up(E_MOL, 256), 256, 0, stream>>>(mdst, E_MOL, DEG);
    deg_to_dinv<<<div_up(N_MOL, 256), 256, 0, stream>>>(DEG, N_MOL);

    gemm(mol_x, GW1T, H1, N_MOL, GCN_H, GCN_IN, 1, 0, 0, 0);
    gcn_self<<<div_up((long long)N_MOL * GCN_H, 256), 256, 0, stream>>>(H1, DEG, AGG1, N_MOL, GCN_H);
    gcn_edge<<<div_up(E_MOL, 8), 256, 0, stream>>>(msrc, mdst, E_MOL, H1, DEG, AGG1, GCN_H);
    bn_relu_seq(AGG1, N_MOL, GCN_H);                                      // G1

    gemm(AGG1, GW2T, H2, N_MOL, DIM_D, GCN_H, 1, 0, 0, 0);
    gcn_self<<<div_up((long long)N_MOL * DIM_D, 256), 256, 0, stream>>>(H2, DEG, AGG2, N_MOL, DIM_D);
    gcn_edge<<<div_up(E_MOL, 8), 256, 0, stream>>>(msrc, mdst, E_MOL, H2, DEG, AGG2, DIM_D);
    bn_relu_seq(AGG2, N_MOL, DIM_D);                                      // G2

    // ===== global mean pool =====
    hipMemsetAsync(GSUM, 0, (size_t)N_DRUGS * DIM_D * sizeof(float), stream);
    hipMemsetAsync(GCNT, 0, (size_t)N_DRUGS * sizeof(float), stream);
    pool_scatter<<<div_up(N_MOL, 8), 256, 0, stream>>>(AGG2, mol_bat, GSUM, GCNT, N_MOL, DIM_D);
    pool_div<<<div_up((long long)N_DRUGS * DIM_D, 256), 256, 0, stream>>>(GSUM, GCNT, N_DRUGS, DIM_D);

    // ===== attention fusion -> X[0:4096], beta -> out[19020:] =====
    dim3 ag(N_DRUGS, 2);
    att_score<<<ag, 128, 0, stream>>>(GSUM, FP, att_w1, att_b1, att_w2, ATTW);
    att_fuse<<<N_DRUGS, DIM_D, 0, stream>>>(GSUM, FP, ATTW, X, out + (size_t)N_KG * 2);
    copy_f32<<<div_up((long long)N_GENE * DIM_D, 256), 256, 0, stream>>>(
        X + (size_t)N_DRUGS * DIM_D, gene_emb, (long long)N_GENE * DIM_D);

    // ===== RGCN layer 1 =====
    gemm(X, RGW1T, HR, N_KG, DIM1, DIM_D, R_REL, 0,
         (long long)DIM_D * DIM1, (long long)N_KG * DIM1);
    hipMemsetAsync(CNT, 0, (size_t)R_REL * N_KG * sizeof(float), stream);
    rgcn_cnt<<<div_up(E_KG, 256), 256, 0, stream>>>(kdst, kg_et, E_KG, CNT);
    hipMemsetAsync(AGGK, 0, (size_t)N_KG * DIM1 * sizeof(float), stream);
    rgcn_edge<<<div_up(E_KG, 8), 256, 0, stream>>>(ksrc, kdst, kg_et, E_KG, HR, CNT, AGGK, DIM1);
    gemm(X, RGR1T, XROOT, N_KG, DIM1, DIM_D, 1, 0, 0, 0);
    vec_add<<<div_up((long long)N_KG * DIM1, 256), 256, 0, stream>>>(AGGK, XROOT, (long long)N_KG * DIM1);
    bn_relu_seq(AGGK, N_KG, DIM1);                                        // X1

    // ===== RGCN layer 2 (same cnt; XROOT region reused) =====
    gemm(AGGK, RGW2T, HR2, N_KG, DIM2, DIM1, R_REL, 0,
         (long long)DIM1 * DIM2, (long long)N_KG * DIM2);
    gemm(AGGK, RGR2T, XROOT2, N_KG, DIM2, DIM1, 1, 0, 0, 0);
    hipMemsetAsync(AGGK2, 0, (size_t)N_KG * DIM2 * sizeof(float), stream);
    rgcn_edge<<<div_up(E_KG, 8), 256, 0, stream>>>(ksrc, kdst, kg_et, E_KG, HR2, CNT, AGGK2, DIM2);
    vec_add<<<div_up((long long)N_KG * DIM2, 256), 256, 0, stream>>>(AGGK2, XROOT2, (long long)N_KG * DIM2);
    bn_relu_seq(AGGK2, N_KG, DIM2);                                       // X2

    // ===== head =====
    gemm(AGGK2, L1WT, X3, N_KG, DIM3, DIM2, 1, 0, 0, 0);
    bn_relu_seq(X3, N_KG, DIM3);
    lin2_logsoftmax<<<div_up(N_KG, 256), 256, 0, stream>>>(X3, lin2_w, lin2_b, out, N_KG);
}
